// Linformer_35888746726058
// MI455X (gfx1250) — matrix operand order repaired and checked by reading, not compiled, not run on hardware
//
#include <hip/hip_runtime.h>
#include <hip/hip_bf16.h>

// ---- problem constants --------------------------------------------------
#define BH   64      // B*H
#define SEQ  4096
#define HD   64      // head dim
#define PRJ  256     // projection dim

// ---- CDNA5 async global->LDS copy support (guarded; fallback = plain) ----
#if defined(__has_builtin)
#if __has_builtin(__builtin_amdgcn_global_load_async_to_lds_b128)
#define HAVE_ASYNC_LDS 1
#endif
#endif

// ---- vector types -------------------------------------------------------
typedef __attribute__((ext_vector_type(16))) __bf16 v16bf;
typedef __attribute__((ext_vector_type(8)))  __bf16 bf16x8;
typedef __attribute__((ext_vector_type(8)))  float  v8f;
typedef __attribute__((ext_vector_type(4)))  float  f32x4;
typedef __attribute__((ext_vector_type(4)))  int    i32x4;

__device__ __forceinline__ v16bf bf_combine(bf16x8 lo, bf16x8 hi) {
    v16bf r;
#pragma unroll
    for (int i = 0; i < 8; ++i) { r[i] = lo[i]; r[i + 8] = hi[i]; }
    return r;
}

__device__ __forceinline__ v8f wmma_bf16(v16bf a, v16bf b, v8f c) {
    return __builtin_amdgcn_wmma_f32_16x16x32_bf16(
        false, a, false, b, (short)0, c, false, false);
}

// copy 16B global -> LDS, per lane
__device__ __forceinline__ void cp_g2l_16B(const __bf16* g, __bf16* l) {
#if defined(HAVE_ASYNC_LDS)
    __builtin_amdgcn_global_load_async_to_lds_b128(
        (__attribute__((address_space(1))) i32x4*)g,
        (__attribute__((address_space(3))) i32x4*)l, 0, 0);
#else
    *(bf16x8*)l = *(const bf16x8*)g;
#endif
}

__device__ __forceinline__ void async_fence() {
#if defined(HAVE_ASYNC_LDS)
#if __has_builtin(__builtin_amdgcn_s_wait_asynccnt)
    __builtin_amdgcn_s_wait_asynccnt(0);
#else
    asm volatile("s_wait_asynccnt 0x0" ::: "memory");
#endif
#endif
}

// =========================================================================
// Kernel 1: low-rank projection.  OUT = EF[P,S] x KV[S,D]  per head.
//   transposed==0 : out layout [bh][P][D]   (KP, for score GEMM B-operand)
//   transposed==1 : out layout [bh][D][P]   (VP^T, for output GEMM B-operand)
// grid = (BH, 4), block = 256 (8 waves). Each block: 64 P-rows x 64 D-cols.
// =========================================================================
__global__ __launch_bounds__(256) void linf_proj_kernel(
    const float* __restrict__ kv,   // [BH,S,D] f32 (key or value)
    const float* __restrict__ ef,   // [P,S]    f32 (E or F)
    __bf16* __restrict__ outp,
    int transposed)
{
    const int bh   = blockIdx.x;
    const int pc   = blockIdx.y;            // 64-row chunk of P
    const int tid  = threadIdx.x;
    const int wave = tid >> 5;
    const int lane = tid & 31;
    const int lh   = lane >> 4;             // half-wave select
    const int l16  = lane & 15;
    const int mt   = wave & 3;              // p tile (16 rows) within chunk
    const int nt0  = (wave >> 2) * 2;       // two d tiles per wave

    // K/V tile staged TRANSPOSED: Bt[d][s_local], padded row stride 40
    __shared__ __bf16 Bt[64][40];

    const float* kvb  = kv + (size_t)bh * SEQ * HD;
    const int    prow = pc * 64 + mt * 16 + l16;    // A-matrix row (p)
    const float* erow = ef + (size_t)prow * SEQ;
    const int    koff = lh * 8;

    v8f acc0 = {}; v8f acc1 = {};

    for (int s0 = 0; s0 < SEQ; s0 += 32) {
        __syncthreads();
        // ---- cooperative stage of kv[s0:s0+32][0:64] into Bt (transposed)
        {
            const int s  = tid >> 4;          // 0..15
            const int d4 = (tid & 15) * 4;    // 0,4,...,60
            f32x4 v0 = *(const f32x4*)(kvb + (size_t)(s0 + s)      * HD + d4);
            f32x4 v1 = *(const f32x4*)(kvb + (size_t)(s0 + s + 16) * HD + d4);
#pragma unroll
            for (int i = 0; i < 4; ++i) {
                Bt[d4 + i][s]      = (__bf16)v0[i];
                Bt[d4 + i][s + 16] = (__bf16)v1[i];
            }
        }
        __syncthreads();

        // ---- A fragment: E/F row, ISA 16-bit A layout, f32->bf16
        v16bf a;
        {
            const float* p0 = erow + s0 + koff;
            const float* p1 = erow + s0 + 16 + koff;
#pragma unroll
            for (int i = 0; i < 8; ++i) a[i]     = (__bf16)p0[i];
#pragma unroll
            for (int i = 0; i < 8; ++i) a[i + 8] = (__bf16)p1[i];
        }

        // ---- B fragments from LDS (conflict-free, contiguous rows of Bt)
        const __bf16* r0 = &Bt[ nt0      * 16 + l16][lh * 8];
        const __bf16* r1 = &Bt[(nt0 + 1) * 16 + l16][lh * 8];
        v16bf b0 = bf_combine(*(const bf16x8*)r0, *(const bf16x8*)(r0 + 16));
        v16bf b1 = bf_combine(*(const bf16x8*)r1, *(const bf16x8*)(r1 + 16));

        acc0 = wmma_bf16(a, b0, acc0);
        acc1 = wmma_bf16(a, b1, acc1);
    }

    // ---- write results (bf16)
    if (!transposed) {
        __bf16* o = outp + (size_t)bh * PRJ * HD;
#pragma unroll
        for (int r = 0; r < 8; ++r) {
            const int p  = pc * 64 + mt * 16 + r + 8 * lh;
            const int d0 = nt0 * 16 + l16;
            o[(size_t)p * HD + d0]      = (__bf16)acc0[r];
            o[(size_t)p * HD + d0 + 16] = (__bf16)acc1[r];
        }
    } else {
        __bf16* o = outp + (size_t)bh * HD * PRJ;
#pragma unroll
        for (int r = 0; r < 8; ++r) {
            const int p  = pc * 64 + mt * 16 + r + 8 * lh;
            const int d0 = nt0 * 16 + l16;
            o[(size_t)d0 * PRJ + p]        = (__bf16)acc0[r];
            o[(size_t)(d0 + 16) * PRJ + p] = (__bf16)acc1[r];
        }
    }
}

// =========================================================================
// Kernel 2: attention.  scores = Q*KP^T/8, softmax over P, out = probs*VP.
// grid = (BH, 32), block = 256 (8 waves); wave owns a 16-row S tile.
// Dynamic LDS layout (bf16 elements):
//   [0 , 16896)            VPT_s  [64][264]            (persistent)
//   [16896, 16896+33792)   SCR: phase 1 = KP_s [256][72]
//                               phase 2 = probs [8][16][264]
// total = 50688 elems = 101376 bytes
// =========================================================================
#define VPT_ELEMS   (64 * 264)
#define KPS_STRIDE  72
#define PRB_STRIDE  264
#define SCR_ELEMS   (8 * 16 * 264)               // >= 256*72
#define ATTN_SMEM_BYTES ((VPT_ELEMS + SCR_ELEMS) * 2)

__global__ __launch_bounds__(256) void linf_attn_kernel(
    const float*  __restrict__ q,    // [BH,S,D] f32
    const __bf16* __restrict__ kp,   // [BH,P,D] bf16
    const __bf16* __restrict__ vpt,  // [BH,D,P] bf16
    float* __restrict__ out)         // [BH,S,D] f32
{
    extern __shared__ __bf16 smem[];
    __bf16* VPTs = smem;                 // [64][264]
    __bf16* KPs  = smem + VPT_ELEMS;     // [256][72]   (phase 1)
    __bf16* PRB  = smem + VPT_ELEMS;     // [8][16][264] (phase 2, overlays KPs)

    const int bh   = blockIdx.x;
    const int tid  = threadIdx.x;
    const int wave = tid >> 5;
    const int lane = tid & 31;
    const int lh   = lane >> 4;
    const int l16  = lane & 15;
    const int koff = lh * 8;
    const int s0   = blockIdx.y * 128 + wave * 16;   // this wave's 16 Q rows

    // ---- cooperative stage of KP [256][64] and VPT [64][256] into LDS
    {
        // KP: thread tid copies one full 64-elem row (128B, coalesced)
        const __bf16* g = kp + (size_t)bh * PRJ * HD + (size_t)tid * HD;
        __bf16*       l = KPs + tid * KPS_STRIDE;
#pragma unroll
        for (int j = 0; j < 8; ++j) cp_g2l_16B(g + j * 8, l + j * 8);
    }
    {
        // VPT: thread copies a 64-elem quarter row
        const int row = tid >> 2, seg = (tid & 3) * 64;
        const __bf16* g = vpt + (size_t)bh * HD * PRJ + (size_t)row * PRJ + seg;
        __bf16*       l = VPTs + row * PRB_STRIDE + seg;
#pragma unroll
        for (int j = 0; j < 8; ++j) cp_g2l_16B(g + j * 8, l + j * 8);
    }

    // ---- Q A-fragments, scaled by 1/sqrt(D)=0.125, f32 -> bf16
    // (overlaps with outstanding async staging)
    const float* qrow = q + (size_t)bh * SEQ * HD + (size_t)(s0 + l16) * HD;
    v16bf aq[2];
#pragma unroll
    for (int kc = 0; kc < 2; ++kc) {
        const float* p0 = qrow + kc * 32 + koff;
        const float* p1 = qrow + kc * 32 + 16 + koff;
#pragma unroll
        for (int i = 0; i < 8; ++i) aq[kc][i]     = (__bf16)(p0[i] * 0.125f);
#pragma unroll
        for (int i = 0; i < 8; ++i) aq[kc][i + 8] = (__bf16)(p1[i] * 0.125f);
    }
    async_fence();       // wave's own async DMAs complete
    __syncthreads();     // all waves' staging visible

    // ---- scores: 16 WMMA tiles over P, K=64 in two 32-chunks (B from LDS)
    v8f c[16];
#pragma unroll
    for (int t = 0; t < 16; ++t) {
        const __bf16* rb = KPs + (size_t)(t * 16 + l16) * KPS_STRIDE;
        v16bf b0 = bf_combine(*(const bf16x8*)(rb + lh * 8),
                              *(const bf16x8*)(rb + lh * 8 + 16));
        v16bf b1 = bf_combine(*(const bf16x8*)(rb + 32 + lh * 8),
                              *(const bf16x8*)(rb + 32 + lh * 8 + 16));
        v8f z = {};
        z    = wmma_bf16(aq[0], b0, z);
        c[t] = wmma_bf16(aq[1], b1, z);
    }

    // ---- in-register softmax across P=256 (per C/D-layout row)
#pragma unroll
    for (int r = 0; r < 8; ++r) {
        float m = c[0][r];
#pragma unroll
        for (int t = 1; t < 16; ++t) m = fmaxf(m, c[t][r]);
#pragma unroll
        for (int off = 1; off < 16; off <<= 1)
            m = fmaxf(m, __shfl_xor(m, off, 32));
        float s = 0.0f;
#pragma unroll
        for (int t = 0; t < 16; ++t) { float e = __expf(c[t][r] - m); c[t][r] = e; s += e; }
#pragma unroll
        for (int off = 1; off < 16; off <<= 1)
            s += __shfl_xor(s, off, 32);
        const float inv = 1.0f / s;
#pragma unroll
        for (int t = 0; t < 16; ++t) c[t][r] *= inv;
    }

    // KPs region is dead for every wave from here on; reuse it for probs.
    __syncthreads();

    // ---- reshape probs into A-fragment layout via LDS
    __bf16* myp = PRB + (size_t)wave * 16 * PRB_STRIDE;
#pragma unroll
    for (int t = 0; t < 16; ++t)
#pragma unroll
        for (int r = 0; r < 8; ++r)
            myp[(size_t)(r + 8 * lh) * PRB_STRIDE + t * 16 + l16] = (__bf16)c[t][r];

    // ---- out = probs[16,256] x VP[256,64]   (both operands from LDS)
    v8f o[4] = {{}, {}, {}, {}};
#pragma unroll
    for (int kc = 0; kc < 8; ++kc) {
        const __bf16* pr0 = myp + (size_t)l16 * PRB_STRIDE + kc * 32 + koff;
        const __bf16* pr1 = pr0 + 16;
        v16bf ap = bf_combine(*(const bf16x8*)pr0, *(const bf16x8*)pr1);
#pragma unroll
        for (int dt = 0; dt < 4; ++dt) {
            const __bf16* vr = VPTs + (size_t)(dt * 16 + l16) * PRB_STRIDE
                             + kc * 32 + lh * 8;
            v16bf b = bf_combine(*(const bf16x8*)vr, *(const bf16x8*)(vr + 16));
            o[dt] = wmma_bf16(ap, b, o[dt]);
        }
    }

    // ---- coalesced f32 store
    float* ob = out + (size_t)bh * SEQ * HD + (size_t)s0 * HD;
#pragma unroll
    for (int dt = 0; dt < 4; ++dt)
#pragma unroll
        for (int r = 0; r < 8; ++r)
            ob[(size_t)(r + 8 * lh) * HD + dt * 16 + l16] = o[dt][r];
}

// =========================================================================
extern "C" void kernel_launch(void* const* d_in, const int* in_sizes, int n_in,
                              void* d_out, int out_size, void* d_ws, size_t ws_size,
                              hipStream_t stream) {
    const float* q = (const float*)d_in[0];
    const float* k = (const float*)d_in[1];
    const float* v = (const float*)d_in[2];
    const float* E = (const float*)d_in[3];
    const float* F = (const float*)d_in[4];

    __bf16* kp  = (__bf16*)d_ws;                        // [BH,P,D]  2 MB
    __bf16* vpt = kp + (size_t)BH * PRJ * HD;           // [BH,D,P]  2 MB

    // opt in to >64KB dynamic LDS (non-stream API: safe under graph capture)
    static int attr_set = 0;
    if (!attr_set) {
        (void)hipFuncSetAttribute((const void*)linf_attn_kernel,
                                  hipFuncAttributeMaxDynamicSharedMemorySize,
                                  ATTN_SMEM_BYTES);
        attr_set = 1;
    }

    linf_proj_kernel<<<dim3(BH, 4), 256, 0, stream>>>(k, E, kp, 0);
    linf_proj_kernel<<<dim3(BH, 4), 256, 0, stream>>>(v, F, vpt, 1);
    linf_attn_kernel<<<dim3(BH, SEQ / 128), 256, ATTN_SMEM_BYTES, stream>>>(
        q, kp, vpt, (float*)d_out);
}
